// Resnet34_VLAD_53317724012673
// MI455X (gfx1250) — compile-verified, matrix operand order checked by reading, not run on hardware
//
#include <hip/hip_runtime.h>
#include <hip/hip_bf16.h>
#include <utility>

// ---------------------------------------------------------------------------
// Types for CDNA5 WMMA (wave32): D(16x16 f32) = A(16x32 bf16) x B(32x16 bf16) + C
// ---------------------------------------------------------------------------
typedef __attribute__((ext_vector_type(16))) __bf16 bf16x16;
typedef __attribute__((ext_vector_type(8)))  __bf16 bf16x8;
typedef __attribute__((ext_vector_type(8)))  float  f32x8;
typedef __attribute__((ext_vector_type(4)))  float  f32x4;
typedef __attribute__((ext_vector_type(4)))  unsigned int u32x4;
typedef __attribute__((ext_vector_type(8)))  int i32x8;
typedef __attribute__((ext_vector_type(4)))  int i32x4;

#if __has_builtin(__builtin_amdgcn_tensor_load_to_lds)
#define HAVE_TDM 1
#else
#define HAVE_TDM 0
#endif

// K-index held by (lane, element e) for the 16-bit 16x32 A / 32x16 B fragment
// layouts (ISA 7.12.2): e0..7 -> K = loff+0..7, e8..15 -> K = loff+16..23,
// with loff = 8 for lanes 16..31.
__device__ __forceinline__ int kmap(int lane, int e) {
  int k = (e < 8) ? e : (8 + e);          // 0..7 -> 0..7, 8..15 -> 16..23
  return k + ((lane & 16) ? 8 : 0);
}

#if HAVE_TDM
// Issue a TDM 2D tile load (32 rows x 32 bf16 cols) from a row-major
// (rows x strideElems) bf16 matrix into LDS at ldsByte. OOB rows/cols beyond
// (rowsValid, colsValid) are zero-filled by the TDM (ISA ch.8 D# semantics).
// This toolchain exposes the 6-arg builtin:
//   (uint32x4 g0, int32x8 g1, int32x4 g2, int32x4 g3, int32x8 extra, int cpol)
__device__ __forceinline__ void tdm_load_wtile(const __bf16* gsrc, unsigned ldsByte,
                                               int rowsValid, int colsValid,
                                               long long strideElems) {
  unsigned long long ga = (unsigned long long)(__UINTPTR_TYPE__)gsrc;
  u32x4 g0;
  g0[0] = 1u;                                   // count=1, user mode, no gather
  g0[1] = ldsByte;                              // lds_addr
  g0[2] = (unsigned)(ga & 0xFFFFFFFFu);         // global_addr[31:0]
  g0[3] = (unsigned)((ga >> 32) & 0x1FFFFFFu)   // global_addr[56:32]
          | (2u << 30);                         // type=2 ("image")
  unsigned T0 = (unsigned)colsValid;            // tensor_dim0 (elements)
  unsigned T1 = (unsigned)rowsValid;            // tensor_dim1 (rows)
  unsigned long long S0 = (unsigned long long)strideElems;
  i32x8 g1;
  g1[0] = (int)(1u << 16);                      // data_size = 1 (2 bytes)
  g1[1] = (int)((T0 & 0xFFFFu) << 16);          // tensor_dim0[15:0]
  g1[2] = (int)((T0 >> 16) | ((T1 & 0xFFFFu) << 16));
  g1[3] = (int)((T1 >> 16) | (32u << 16));      // tile_dim0 = 32
  g1[4] = (int)32u;                             // tile_dim1 = 32, tile_dim2 = 0
  g1[5] = (int)(unsigned)(S0 & 0xFFFFFFFFull);  // tensor_dim0_stride[31:0]
  g1[6] = (int)(unsigned)((S0 >> 32) & 0xFFFFull);
  g1[7] = 0;
  i32x4 gz4 = {0, 0, 0, 0};                     // 2-D tensor: groups 2/3 unused
  i32x8 gz8 = {0, 0, 0, 0, 0, 0, 0, 0};
  __builtin_amdgcn_tensor_load_to_lds(g0, g1, gz4, gz4, gz8, 0);
}
#endif

// ---------------------------------------------------------------------------
// Implicit-GEMM convolution with fused BN / bias / residual / ReLU epilogue.
// Output GEMM view: O[co, m] with m = (n, ho, wo), K = Cin*KH*KW.
// Block = 4 waves; block tile = 32 (Cout) x 64 (m). Weight tile (32x32 bf16)
// staged to LDS by the Tensor Data Mover; each wave reuses its B fragment for
// two WMMAs (two 16-row Cout sub-tiles).
// ---------------------------------------------------------------------------
struct ConvCfg {
  const float* in;
  const __bf16* wbf;       // (Cout, Cin*KH*KW) bf16, preconverted
  const float* g;          // BN gamma
  const float* bt;         // BN beta
  const float* mu;         // BN mean
  const float* var;        // BN var
  const float* bias;       // bias mode
  const float* residual;   // optional, same shape as out
  float* out;
  int N, Cin, Hin, Win;
  int Cout, Hout, Wout;
  int stride, pad;
  int ostride;             // output row stride (>= Wout; allows padded outputs)
  int epil;                // 0 = BN, 1 = bias
  int relu;                // applied after residual add
};

template <int KH, int KW>
__global__ __launch_bounds__(128) void conv_wmma(ConvCfg cfg) {
  const int lane = threadIdx.x & 31;
  const int wave = threadIdx.x >> 5;
  const int rowBase = blockIdx.y * 32;
  const int Ncols = cfg.N * cfg.Hout * cfg.Wout;
  const int colBase = blockIdx.x * 64 + wave * 16;
  const int K = cfg.Cin * KH * KW;
  constexpr int KHKW = KH * KW;

  __shared__ __align__(16) __bf16 lA[32 * 32];  // [row][k] weight tile

  // This lane's output column for the B fragment
  const int c = colBase + (lane & 15);
  const bool cvalid = (c < Ncols);
  int wo = 0, ho = 0, nb = 0;
  if (cvalid) {
    wo = c % cfg.Wout;
    int t = c / cfg.Wout;
    ho = t % cfg.Hout;
    nb = t / cfg.Hout;
  }
  const int hw = cfg.Hin * cfg.Win;
  const int hi0 = ho * cfg.stride - cfg.pad;
  const int wi0 = wo * cfg.stride - cfg.pad;
  // Whole receptive field in-bounds? (always true for 1x1, pad==0)
  const bool interior =
      (KHKW == 1) ? cvalid
                  : (cvalid && hi0 >= 0 && (hi0 + KH) <= cfg.Hin &&
                     wi0 >= 0 && (wi0 + KW) <= cfg.Win);
  const int loff = (lane & 16) ? 8 : 0;

  f32x8 acc0 = {}, acc1 = {};
  for (int k0 = 0; k0 < K; k0 += 32) {
    __syncthreads();   // previous tile fully consumed
#if HAVE_TDM
    if (wave == 0) {
      unsigned ldsByte = (unsigned)(__UINTPTR_TYPE__)(&lA[0]);
      tdm_load_wtile(cfg.wbf + (size_t)rowBase * K + k0, ldsByte,
                     cfg.Cout - rowBase, K - k0, (long long)K);
      __builtin_amdgcn_s_wait_tensorcnt(0);
    }
#else
    for (int idx = threadIdx.x; idx < 32 * 32; idx += 128) {
      int m = idx >> 5, kk = idx & 31;
      int co = rowBase + m;
      int kg = k0 + kk;
      lA[idx] = (co < cfg.Cout && kg < K) ? cfg.wbf[(size_t)co * K + kg]
                                          : (__bf16)0.0f;
    }
#endif
    __syncthreads();   // tile published

    // A fragments: two 16-byte LDS reads per 16-row sub-tile
    const __bf16* row0 = &lA[((lane & 15) << 5) + loff];
    const __bf16* row1 = &lA[(((lane & 15) + 16) << 5) + loff];
    bf16x8 a0lo = *(const bf16x8*)(row0);
    bf16x8 a0hi = *(const bf16x8*)(row0 + 16);
    bf16x8 a1lo = *(const bf16x8*)(row1);
    bf16x8 a1hi = *(const bf16x8*)(row1 + 16);
    bf16x16 a0, a1;
#pragma unroll
    for (int e = 0; e < 8; ++e) {
      a0[e] = a0lo[e]; a0[e + 8] = a0hi[e];
      a1[e] = a1lo[e]; a1[e + 8] = a1hi[e];
    }

    // B fragment: branch-free gather when interior and full K-step
    bf16x16 b;
    if (interior && (k0 + 32 <= K)) {
      const float* basep =
          cfg.in + (size_t)nb * cfg.Cin * hw + (size_t)hi0 * cfg.Win + wi0;
      float tmp[16];
#pragma unroll
      for (int e = 0; e < 16; ++e) {
        const int kk = kmap(lane, e);
        size_t o;
        if (KHKW == 1) {
          o = (size_t)(k0 + kk) * hw;
        } else {
          const int kg = k0 + kk;
          const int ci = kg / KHKW;
          const int r = kg - ci * KHKW;
          const int dy = (KW == 1) ? r : r / KW;
          const int dx = (KW == 1) ? 0 : r - dy * KW;
          o = (size_t)ci * hw + (size_t)dy * cfg.Win + dx;
        }
        tmp[e] = basep[o];
      }
      if (KHKW == 1) __builtin_prefetch(basep + (size_t)(k0 + 32) * hw, 0, 1);
#pragma unroll
      for (int e = 0; e < 16; ++e) b[e] = (__bf16)tmp[e];
    } else {
#pragma unroll
      for (int e = 0; e < 16; ++e) {
        const int kk = kmap(lane, e);
        const int kg = k0 + kk;
        float vb = 0.0f;
        if (cvalid && kg < K) {
          const int ci = (KHKW == 1) ? kg : kg / KHKW;
          const int r = kg - ci * KHKW;
          const int dy = (KW == 1) ? r : r / KW;
          const int dx = (KW == 1) ? 0 : r - dy * KW;
          const int hi = hi0 + dy;
          const int wi = wi0 + dx;
          if (hi >= 0 && hi < cfg.Hin && wi >= 0 && wi < cfg.Win)
            vb = cfg.in[(((size_t)nb * cfg.Cin + ci) * cfg.Hin + hi) * cfg.Win + wi];
        }
        b[e] = (__bf16)vb;
      }
    }

    acc0 = __builtin_amdgcn_wmma_f32_16x16x32_bf16(false, a0, false, b,
                                                   (short)0, acc0, false, false);
    acc1 = __builtin_amdgcn_wmma_f32_16x16x32_bf16(false, a1, false, b,
                                                   (short)0, acc1, false, false);
  }

  // Epilogue: C/D layout — VGPR r: m = r + 8*(lane>=16), n = lane & 15.
  // The output column (nb, ho, wo) was already derived for this lane.
  if (cvalid) {
#pragma unroll
    for (int sub = 0; sub < 2; ++sub) {
      const f32x8& acc = sub ? acc1 : acc0;
#pragma unroll
      for (int r = 0; r < 8; ++r) {
        const int m  = r + ((lane & 16) ? 8 : 0);
        const int co = rowBase + sub * 16 + m;
        if (co >= cfg.Cout) continue;
        float y = acc[r];
        if (cfg.epil == 0) {
          float s = cfg.g[co] * rsqrtf(cfg.var[co] + 1e-5f);
          y = (y - cfg.mu[co]) * s + cfg.bt[co];
        } else {
          y += cfg.bias[co];
        }
        size_t oi = (((size_t)nb * cfg.Cout + co) * cfg.Hout + ho) * cfg.ostride + wo;
        if (cfg.residual) y += cfg.residual[oi];
        if (cfg.relu) y = fmaxf(y, 0.0f);
        cfg.out[oi] = y;
      }
    }
  }
}

// float -> bf16 weight conversion (run once per layer; weights are L2-resident)
__global__ void cvt_bf16(const float* __restrict__ w, __bf16* __restrict__ o, int n) {
  int i = blockIdx.x * blockDim.x + threadIdx.x;
  if (i < n) o[i] = (__bf16)w[i];
}

__global__ void zero_f32(float* __restrict__ p, int n) {
  int i = blockIdx.x * blockDim.x + threadIdx.x;
  if (i < n) p[i] = 0.0f;
}

// ---------------------------------------------------------------------------
// conv1 (Cin=1, 7x7 pad3) + BN + ReLU + 2x2/2 maxpool, fused (no GEMM benefit)
// ---------------------------------------------------------------------------
__global__ void conv1_bn_relu_pool(const float* __restrict__ x,
                                   const float* __restrict__ w,
                                   const float* __restrict__ g,
                                   const float* __restrict__ bt,
                                   const float* __restrict__ mu,
                                   const float* __restrict__ var,
                                   float* __restrict__ out,
                                   int N, int Hin, int Win, int Hp, int Wp) {
  size_t i = (size_t)blockIdx.x * blockDim.x + threadIdx.x;
  size_t total = (size_t)N * 64 * Hp * Wp;
  if (i >= total) return;
  int wp = i % Wp;
  size_t t = i / Wp;
  int hp = t % Hp; t /= Hp;
  int co = t % 64;
  int nb = (int)(t / 64);
  float s = g[co] * rsqrtf(var[co] + 1e-5f);
  float best = -__builtin_inff();
  for (int py = 0; py < 2; ++py) {
    for (int px = 0; px < 2; ++px) {
      int ho = hp * 2 + py, wo = wp * 2 + px;
      float a = 0.0f;
      for (int dy = 0; dy < 7; ++dy) {
        int hi = ho - 3 + dy;
        if (hi < 0 || hi >= Hin) continue;
        for (int dx = 0; dx < 7; ++dx) {
          int wi = wo - 3 + dx;
          if (wi < 0 || wi >= Win) continue;
          a += x[((size_t)nb * Hin + hi) * Win + wi] * w[(co * 7 + dy) * 7 + dx];
        }
      }
      float y = fmaxf((a - mu[co]) * s + bt[co], 0.0f);
      best = fmaxf(best, y);
    }
  }
  out[i] = best;
}

// (3,1) stride (2,1) max pool
__global__ void maxpool31(const float* __restrict__ in, float* __restrict__ out,
                          int N, int C, int Hin, int Win, int Hout) {
  size_t i = (size_t)blockIdx.x * blockDim.x + threadIdx.x;
  size_t total = (size_t)N * C * Hout * Win;
  if (i >= total) return;
  int w = i % Win;
  size_t t = i / Win;
  int ho = t % Hout; t /= Hout;
  int c = t % C;
  int nb = (int)(t / C);
  float m = -__builtin_inff();
  for (int d = 0; d < 3; ++d)
    m = fmaxf(m, in[(((size_t)nb * C + c) * Hin + ho * 2 + d) * Win + w]);
  out[i] = m;
}

// softmax over the 12 cluster scores; scores (N,12,1,W) -> At (N,16,64) padded,
// rows k>=12 and columns t>=W stay zero (buffer pre-zeroed).
__global__ void softmax12(const float* __restrict__ scores, float* __restrict__ At,
                          int N, int W) {
  int i = blockIdx.x * blockDim.x + threadIdx.x;
  if (i >= N * W) return;
  int t = i % W, nb = i / W;
  float v[12];
  float mx = -__builtin_inff();
  for (int k = 0; k < 12; ++k) {
    v[k] = scores[((size_t)nb * 12 + k) * W + t];
    mx = fmaxf(mx, v[k]);
  }
  float s = 0.0f;
  for (int k = 0; k < 12; ++k) { v[k] = __expf(v[k] - mx); s += v[k]; }
  float inv = 1.0f / s;
  for (int k = 0; k < 12; ++k) At[((size_t)nb * 16 + k) * 64 + t] = v[k] * inv;
}

__global__ void asum_k(const float* __restrict__ At, float* __restrict__ asum, int N) {
  int i = blockIdx.x * blockDim.x + threadIdx.x;
  if (i >= N * 12) return;
  int k = i % 12, nb = i / 12;
  float s = 0.0f;
  for (int t = 0; t < 64; ++t) s += At[((size_t)nb * 16 + k) * 64 + t];  // pad = 0
  asum[i] = s;
}

// ---------------------------------------------------------------------------
// VLAD aggregation: cres[b,k,d] = sum_t At[b,k,t]*featP[b,d,t] - asum[b,k]*cluster[k,d]
// One wave per (b, 16-wide d-tile); 16x16x64 via two branch-free WMMA steps.
// At is (8,16,64) and featP is (8*512,64), both zero-padded -> unguarded
// 128-bit loads matching the fragment layout.
// ---------------------------------------------------------------------------
__global__ __launch_bounds__(128) void vlad_wmma(const float* __restrict__ At,
                                                 const float* __restrict__ featP,
                                                 const float* __restrict__ asum,
                                                 const float* __restrict__ cluster,
                                                 float* __restrict__ cres) {
  const int lane = threadIdx.x & 31;
  const int wave = threadIdx.x >> 5;
  const int wid = blockIdx.x * 4 + wave;   // 0..255
  const int b = wid >> 5;                  // 8 batches
  const int dbase = (wid & 31) * 16;       // 512 / 16 tiles
  const int loff = (lane & 16) ? 8 : 0;
  const int n16 = lane & 15;               // A row / B column index

  f32x8 acc = {};
#pragma unroll
  for (int k0 = 0; k0 < 64; k0 += 32) {
    const float* ap = At + ((size_t)b * 16 + n16) * 64 + k0 + loff;
    const float* bp = featP + ((size_t)b * 512 + dbase + n16) * 64 + k0 + loff;
    f32x4 xa0 = *(const f32x4*)(ap);
    f32x4 xa1 = *(const f32x4*)(ap + 4);
    f32x4 xa2 = *(const f32x4*)(ap + 16);
    f32x4 xa3 = *(const f32x4*)(ap + 20);
    f32x4 xb0 = *(const f32x4*)(bp);
    f32x4 xb1 = *(const f32x4*)(bp + 4);
    f32x4 xb2 = *(const f32x4*)(bp + 16);
    f32x4 xb3 = *(const f32x4*)(bp + 20);
    bf16x16 a, bm;
#pragma unroll
    for (int e = 0; e < 4; ++e) {
      a[e] = (__bf16)xa0[e];       a[e + 4]  = (__bf16)xa1[e];
      a[e + 8] = (__bf16)xa2[e];   a[e + 12] = (__bf16)xa3[e];
      bm[e] = (__bf16)xb0[e];      bm[e + 4]  = (__bf16)xb1[e];
      bm[e + 8] = (__bf16)xb2[e];  bm[e + 12] = (__bf16)xb3[e];
    }
    acc = __builtin_amdgcn_wmma_f32_16x16x32_bf16(false, a, false, bm,
                                                  (short)0, acc, false, false);
  }
#pragma unroll
  for (int r = 0; r < 8; ++r) {
    const int kr = r + ((lane & 16) ? 8 : 0);
    const int d = dbase + n16;
    if (kr < 12)
      cres[((size_t)b * 12 + kr) * 512 + d] =
          acc[r] - asum[b * 12 + kr] * cluster[(size_t)kr * 512 + d];
  }
}

// L2-normalize each (b, k<10) 512-vector; one wave per (b,k)
__global__ void l2norm(const float* __restrict__ cres, float* __restrict__ flat) {
  const int bk = blockIdx.x;          // 0..79
  const int b = bk / 10, k = bk % 10;
  const int lane = threadIdx.x;
  const float* src = cres + ((size_t)b * 12 + k) * 512;
  float s = 0.0f;
  for (int d = lane; d < 512; d += 32) { float v = src[d]; s += v * v; }
  for (int off = 16; off > 0; off >>= 1) s += __shfl_xor(s, off, 32);
  float inv = 1.0f / fmaxf(sqrtf(s), 1e-12f);
  for (int d = lane; d < 512; d += 32)
    flat[((size_t)b * 10 + k) * 512 + d] = src[d] * inv;
}

// Dense head: out[b,p] = flat[b,:] . dense_w[p,:] + bias[p]; 16x16x32 WMMA,
// K = 5120, one wave per 16-wide p tile. Per-lane fragments are contiguous
// float runs -> 128-bit global loads.
__global__ __launch_bounds__(128) void dense_wmma(const float* __restrict__ flat,
                                                  const float* __restrict__ Wd,
                                                  const float* __restrict__ bias,
                                                  float* __restrict__ out) {
  const int lane = threadIdx.x & 31;
  const int wave = threadIdx.x >> 5;
  const int pbase = (blockIdx.x * 4 + wave) * 16;   // 32 tiles over 512
  const int loff = (lane & 16) ? 8 : 0;
  const int m = lane & 15;
  const int p = pbase + m;
  f32x8 acc = {};
  for (int k0 = 0; k0 < 5120; k0 += 32) {
    bf16x16 a, bm;
    // B: weights, always valid
    {
      const float* bp = Wd + (size_t)p * 5120 + k0 + loff;
      f32x4 w0 = *(const f32x4*)(bp);
      f32x4 w1 = *(const f32x4*)(bp + 4);
      f32x4 w2 = *(const f32x4*)(bp + 16);
      f32x4 w3 = *(const f32x4*)(bp + 20);
#pragma unroll
      for (int e = 0; e < 4; ++e) {
        bm[e] = (__bf16)w0[e];      bm[e + 4]  = (__bf16)w1[e];
        bm[e + 8] = (__bf16)w2[e];  bm[e + 12] = (__bf16)w3[e];
      }
    }
    // A: activations, rows >= 8 are zero padding
    if (m < 8) {
      const float* ap = flat + (size_t)m * 5120 + k0 + loff;
      f32x4 x0 = *(const f32x4*)(ap);
      f32x4 x1 = *(const f32x4*)(ap + 4);
      f32x4 x2 = *(const f32x4*)(ap + 16);
      f32x4 x3 = *(const f32x4*)(ap + 20);
#pragma unroll
      for (int e = 0; e < 4; ++e) {
        a[e] = (__bf16)x0[e];      a[e + 4]  = (__bf16)x1[e];
        a[e + 8] = (__bf16)x2[e];  a[e + 12] = (__bf16)x3[e];
      }
    } else {
#pragma unroll
      for (int e = 0; e < 16; ++e) a[e] = (__bf16)0.0f;
    }
    acc = __builtin_amdgcn_wmma_f32_16x16x32_bf16(false, a, false, bm,
                                                  (short)0, acc, false, false);
  }
#pragma unroll
  for (int r = 0; r < 8; ++r) {
    const int mm = r + ((lane & 16) ? 8 : 0);
    const int pp = pbase + (lane & 15);
    if (mm < 8) out[(size_t)mm * 512 + pp] = acc[r] + bias[pp];
  }
}

// ---------------------------------------------------------------------------
// Host orchestration
// ---------------------------------------------------------------------------
extern "C" void kernel_launch(void* const* d_in, const int* in_sizes, int n_in,
                              void* d_out, int out_size, void* d_ws, size_t ws_size,
                              hipStream_t stream) {
  (void)in_sizes; (void)n_in; (void)out_size; (void)ws_size;

  // ---- unpack inputs in setup_inputs() dict order ----
  int cur = 0;
  auto nxt = [&]() { return (const float*)d_in[cur++]; };
  struct CB { const float* w; const float* bn[4]; };
  auto nxtCB = [&]() { CB c; c.w = nxt(); for (int t = 0; t < 4; ++t) c.bn[t] = nxt(); return c; };

  const float* x = nxt();
  CB c1; c1.w = nxt(); for (int t = 0; t < 4; ++t) c1.bn[t] = nxt();
  CB cb[4][3]; for (int i = 0; i < 4; ++i) for (int j = 0; j < 3; ++j) cb[i][j] = nxtCB();
  CB sc[4];    for (int i = 0; i < 4; ++i) sc[i] = nxtCB();
  CB ib[4][3]; for (int i = 0; i < 4; ++i) for (int j = 0; j < 3; ++j) ib[i][j] = nxtCB();
  const float* conv2_w = nxt();
  const float* conv2_b = nxt();
  const float* conv3_w = nxt();
  const float* conv3_b = nxt();
  const float* cluster = nxt();
  const float* dense_w = nxt();
  const float* dense_b = nxt();

  // ---- workspace carve-out (floats) ----
  float* ws = (float*)d_ws;
  size_t off = 0;
  auto alloc = [&](size_t n) { float* p = ws + off; off += n; return p; };
  float* bufA = alloc(29491200);           // max stage activation (8,96,128,300)
  float* bufB = alloc(29491200);
  float* scb  = alloc(29491200);
  float* t1   = alloc(14745600);
  float* t2   = alloc(14745600);
  float* hp   = alloc((size_t)8 * 512 * 7 * 38);
  float* featP = alloc((size_t)8 * 512 * 64);  // padded t-dim (38 -> 64)
  float* scr  = alloc((size_t)8 * 12 * 38);
  float* At   = alloc((size_t)8 * 16 * 64);    // padded softmax output
  float* asum = alloc(8 * 12);
  float* cres = alloc((size_t)8 * 12 * 512);
  float* flat = alloc((size_t)8 * 10 * 512);
  __bf16* wbf = (__bf16*)alloc(1024 * 1024);   // bf16 weight scratch (max 1.84M elems)

  auto cdiv = [](int a, int b) { return (a + b - 1) / b; };

  auto conv = [&](const float* in, const CB& p, const float* bias,
                  const float* res, float* out,
                  int N, int Cin, int Hin, int Win, int Cout, int Hout, int Wout,
                  int kh, int kw, int stride, int pad, int ostride,
                  int epil, int relu) {
    int K = Cin * kh * kw;
    int nW = Cout * K;
    cvt_bf16<<<cdiv(nW, 256), 256, 0, stream>>>(p.w, wbf, nW);
    ConvCfg c;
    c.in = in; c.wbf = wbf;
    c.g = p.bn[0]; c.bt = p.bn[1]; c.mu = p.bn[2]; c.var = p.bn[3];
    c.bias = bias; c.residual = res; c.out = out;
    c.N = N; c.Cin = Cin; c.Hin = Hin; c.Win = Win;
    c.Cout = Cout; c.Hout = Hout; c.Wout = Wout;
    c.stride = stride; c.pad = pad; c.ostride = ostride;
    c.epil = epil; c.relu = relu;
    dim3 grid(cdiv(N * Hout * Wout, 64), cdiv(Cout, 32));
    if (kh == 1 && kw == 1)      conv_wmma<1, 1><<<grid, 128, 0, stream>>>(c);
    else if (kh == 3 && kw == 3) conv_wmma<3, 3><<<grid, 128, 0, stream>>>(c);
    else                         conv_wmma<7, 1><<<grid, 128, 0, stream>>>(c);
  };

  const int N = 8;
  // conv1 + bn + relu + 2x2 pool: (8,1,257,600) -> (8,64,128,300)
  {
    size_t total = (size_t)N * 64 * 128 * 300;
    conv1_bn_relu_pool<<<(int)((total + 255) / 256), 256, 0, stream>>>(
        x, c1.w, c1.bn[0], c1.bn[1], c1.bn[2], c1.bn[3], bufA, N, 257, 600, 128, 300);
  }

  // ---- 4 residual stages ----
  const int F[4][3] = {{48, 48, 96}, {96, 96, 128}, {128, 128, 256}, {256, 256, 512}};
  const int strides[4] = {1, 2, 2, 2};
  const int reps[4] = {1, 2, 2, 2};
  int H = 128, W = 300, C = 64;
  float* curB = bufA;
  float* nxtB = bufB;

  for (int s = 0; s < 4; ++s) {
    int st = strides[s];
    int Ho = (st == 1) ? H : (H - 1) / 2 + 1;
    int Wo = (st == 1) ? W : (W - 1) / 2 + 1;
    int f0 = F[s][0], f1 = F[s][1], f2 = F[s][2];

    // conv block + projection shortcut
    conv(curB, cb[s][0], nullptr, nullptr, t1, N, C, H, W, f0, Ho, Wo, 1, 1, st, 0, Wo, 0, 1);
    conv(t1,   cb[s][1], nullptr, nullptr, t2, N, f0, Ho, Wo, f1, Ho, Wo, 3, 3, 1, 1, Wo, 0, 1);
    conv(curB, sc[s],    nullptr, nullptr, scb, N, C, H, W, f2, Ho, Wo, 1, 1, st, 0, Wo, 0, 0);
    conv(t2,   cb[s][2], nullptr, scb,     nxtB, N, f1, Ho, Wo, f2, Ho, Wo, 1, 1, 1, 0, Wo, 0, 1);
    std::swap(curB, nxtB);
    C = f2; H = Ho; W = Wo;

    // identity blocks
    for (int r = 0; r < reps[s]; ++r) {
      conv(curB, ib[s][0], nullptr, nullptr, t1, N, C, H, W, f0, H, W, 1, 1, 1, 0, W, 0, 1);
      conv(t1,   ib[s][1], nullptr, nullptr, t2, N, f0, H, W, f1, H, W, 3, 3, 1, 1, W, 0, 1);
      conv(t2,   ib[s][2], nullptr, curB,    nxtB, N, f1, H, W, f2, H, W, 1, 1, 1, 0, W, 0, 1);
      std::swap(curB, nxtB);
    }
  }
  // now curB = (8, 512, 16, 38)

  // (3,1)/(2,1) maxpool -> (8,512,7,38)
  {
    size_t total = (size_t)N * 512 * 7 * 38;
    maxpool31<<<(int)((total + 255) / 256), 256, 0, stream>>>(curB, hp, N, 512, 16, 38, 7);
  }

  // conv2 (512,512,7,1)+bias+relu -> featP (8,512,1,64 padded);
  // conv3 -> scores (8,12,1,38)
  zero_f32<<<cdiv(8 * 512 * 64, 256), 256, 0, stream>>>(featP, 8 * 512 * 64);
  zero_f32<<<cdiv(8 * 16 * 64, 256), 256, 0, stream>>>(At, 8 * 16 * 64);
  CB pc2; pc2.w = conv2_w; pc2.bn[0] = pc2.bn[1] = pc2.bn[2] = pc2.bn[3] = nullptr;
  CB pc3; pc3.w = conv3_w; pc3.bn[0] = pc3.bn[1] = pc3.bn[2] = pc3.bn[3] = nullptr;
  conv(hp, pc2, conv2_b, nullptr, featP, N, 512, 7, 38, 512, 1, 38, 7, 1, 1, 0, 64, 1, 1);
  conv(hp, pc3, conv3_b, nullptr, scr,   N, 512, 7, 38, 12,  1, 38, 7, 1, 1, 0, 38, 1, 0);

  // softmax over clusters (padded At), per-(b,k) sums of A
  softmax12<<<cdiv(N * 38, 64), 64, 0, stream>>>(scr, At, N, 38);
  asum_k<<<1, 128, 0, stream>>>(At, asum, N);

  // VLAD aggregation (WMMA), L2 norm, dense head (WMMA)
  vlad_wmma<<<64, 128, 0, stream>>>(At, featP, asum, cluster, cres);
  l2norm<<<80, 32, 0, stream>>>(cres, flat);
  dense_wmma<<<8, 128, 0, stream>>>(flat, dense_w, dense_b, (float*)d_out);
}
